// LightningAttention2_18940805775431
// MI455X (gfx1250) — 16-bit and 8-bit matrix operand order repaired and checked by reading, compiled, not run on hardware
//
#include <hip/hip_runtime.h>
#include <hip/hip_bf16.h>
#include <stdint.h>

typedef _Float16 v16h __attribute__((ext_vector_type(16)));
typedef float    v8f  __attribute__((ext_vector_type(8)));
typedef int      v8i  __attribute__((ext_vector_type(8)));

#define BB 2
#define SEQ 8192
#define DIMC 1024
#define NH 16
#define HD 64
#define CH 128
#define TCH 64
#define MTOT (BB*SEQ)
#define NQKV 3072
#define RMS_EPS_BIT 1.1920929e-07f
#define RMS_EPS_OUT 1e-06f
#define NEG_INF (-3.0e38f)

#if defined(__HIP_DEVICE_COMPILE__) && \
    __has_builtin(__builtin_amdgcn_global_load_async_to_lds_b128) && \
    __has_builtin(__builtin_amdgcn_s_wait_asynccnt)
#define HAVE_ASYNC_LDS 1
#else
#define HAVE_ASYNC_LDS 0
#endif

#if HAVE_ASYNC_LDS
typedef int v4i_b128 __attribute__((vector_size(16)));
typedef __attribute__((address_space(1))) v4i_b128 gvec_t;
typedef __attribute__((address_space(3))) v4i_b128 lvec_t;

__device__ __forceinline__ void async_copy_b128(void* lds, const void* gsrc) {
  __builtin_amdgcn_global_load_async_to_lds_b128(
      (gvec_t*)(unsigned long long)(uintptr_t)gsrc,
      (lvec_t*)(unsigned)(uintptr_t)lds,
      0, 0);
}
__device__ __forceinline__ void async_wait0() {
  __builtin_amdgcn_s_wait_asynccnt(0);
}
#else
__device__ __forceinline__ void async_copy_b128(void* lds, const void* gsrc) {
  *(uint4*)lds = *(const uint4*)gsrc;
}
__device__ __forceinline__ void async_wait0() {}
#endif

__device__ __forceinline__ v16h load_a_rowmajor(const _Float16* base, int m0, int ks,
                                                int ldk, int l16, int hi) {
  union { v16h h; unsigned u[8]; } r;
  const _Float16* rowp = base + (size_t)(m0 + l16) * ldk;
#pragma unroll
  for (int v = 0; v < 8; ++v) {
    int K = ks + ((v >> 2) << 4) + ((v & 3) << 1) + (hi << 3);
    r.u[v] = *(const unsigned*)(rowp + K);
  }
  return r.h;
}

__device__ __forceinline__ v16h load_a_colmajor(const _Float16* base, int m0, int ks,
                                                int ldm, int l16, int hi) {
  union { v16h h; _Float16 e[16]; } r;
  int m = m0 + l16;
#pragma unroll
  for (int v = 0; v < 8; ++v) {
    int K = ks + ((v >> 2) << 4) + ((v & 3) << 1) + (hi << 3);
    r.e[2 * v]     = base[(size_t)K * ldm + m];
    r.e[2 * v + 1] = base[(size_t)(K + 1) * ldm + m];
  }
  return r.h;
}

__device__ __forceinline__ v16h load_b_rows(const _Float16* base, int n0, int ks,
                                            int ldn, int l16, int hi) {
  union { v16h h; _Float16 e[16]; } r;
  int n = n0 + l16;
#pragma unroll
  for (int v = 0; v < 8; ++v) {
    int K = ks + ((v >> 2) << 4) + ((v & 3) << 1) + (hi << 3);
    r.e[2 * v]     = base[(size_t)K * ldn + n];
    r.e[2 * v + 1] = base[(size_t)(K + 1) * ldn + n];
  }
  return r.h;
}

__device__ __forceinline__ v16h load_b_transposed(const _Float16* base, int n0, int ks,
                                                  int ldk, int l16, int hi) {
  union { v16h h; unsigned u[8]; } r;
  const _Float16* rowp = base + (size_t)(n0 + l16) * ldk;
#pragma unroll
  for (int v = 0; v < 8; ++v) {
    int K = ks + ((v >> 2) << 4) + ((v & 3) << 1) + (hi << 3);
    r.u[v] = *(const unsigned*)(rowp + K);
  }
  return r.h;
}

__global__ void abssum_kernel(const float* __restrict__ w, size_t n, float* __restrict__ out) {
  float s = 0.f;
  for (size_t i = (size_t)blockIdx.x * blockDim.x + threadIdx.x; i < n;
       i += (size_t)gridDim.x * blockDim.x)
    s += fabsf(w[i]);
#pragma unroll
  for (int off = 16; off; off >>= 1) s += __shfl_down(s, off, 32);
  if ((threadIdx.x & 31) == 0) atomicAdd(out, s);
}

__global__ void wquant_kernel(const float* __restrict__ w, size_t n,
                              const float* __restrict__ sum, float invcount,
                              signed char* __restrict__ wq) {
  float mean  = fmaxf(*sum * invcount, 1e-5f);
  float scale = 1.f / mean;
  for (size_t i = (size_t)blockIdx.x * blockDim.x + threadIdx.x; i < n;
       i += (size_t)gridDim.x * blockDim.x) {
    float q = rintf(w[i] * scale);
    q = fminf(fmaxf(q, -1.f), 1.f);
    wq[i] = (signed char)(int)q;
  }
}

__device__ __forceinline__ float block_red(float v, float* sbuf, int is_max) {
  int lane = threadIdx.x & 31, wv = threadIdx.x >> 5;
#pragma unroll
  for (int off = 16; off; off >>= 1) {
    float o = __shfl_xor(v, off, 32);
    v = is_max ? fmaxf(v, o) : (v + o);
  }
  if (lane == 0) sbuf[wv] = v;
  __syncthreads();
  float r = sbuf[0];
#pragma unroll
  for (int i = 1; i < 8; ++i) r = is_max ? fmaxf(r, sbuf[i]) : (r + sbuf[i]);
  __syncthreads();
  return r;
}

__global__ void act_quant_kernel(const float* __restrict__ x,
                                 const float* __restrict__ nw1, float eps1,
                                 const float* __restrict__ nw2, float eps2,
                                 signed char* __restrict__ xq, float* __restrict__ rs) {
  __shared__ float sbuf[8];
  const int row = blockIdx.x;
  const int tid = threadIdx.x;
  const float* xr = x + (size_t)row * DIMC;

  float v[4], y[4];
  float ss = 0.f;
#pragma unroll
  for (int i = 0; i < 4; ++i) { v[i] = xr[tid + i * 256]; ss += v[i] * v[i]; }
  ss = block_red(ss, sbuf, 0);
  float inv1 = rsqrtf(ss * (1.f / DIMC) + eps1);
#pragma unroll
  for (int i = 0; i < 4; ++i) y[i] = v[i] * inv1 * nw1[tid + i * 256];

  if (nw2) {
    float ss2 = 0.f;
#pragma unroll
    for (int i = 0; i < 4; ++i) ss2 += y[i] * y[i];
    ss2 = block_red(ss2, sbuf, 0);
    float inv2 = rsqrtf(ss2 * (1.f / DIMC) + eps2);
#pragma unroll
    for (int i = 0; i < 4; ++i) y[i] = y[i] * inv2 * nw2[tid + i * 256];
  }

  float amax = 0.f;
#pragma unroll
  for (int i = 0; i < 4; ++i) amax = fmaxf(amax, fabsf(y[i]));
  amax = block_red(amax, sbuf, 1);
  float cl = fmaxf(amax, 1e-5f);
  float s  = 127.f / cl;
#pragma unroll
  for (int i = 0; i < 4; ++i) {
    float q = rintf(y[i] * s);
    q = fminf(fmaxf(q, -128.f), 127.f);
    xq[(size_t)row * DIMC + tid + i * 256] = (signed char)(int)q;
  }
  if (tid == 0) rs[row] = cl * (1.f / 127.f);
}

__global__ void gemm_iu8_kernel(const signed char* __restrict__ Aq,
                                const float* __restrict__ rowscale,
                                const signed char* __restrict__ Wq,
                                const float* __restrict__ wsum, float invcount,
                                const float* __restrict__ bias, int Ncols,
                                float* __restrict__ outF, _Float16* __restrict__ outH) {
  __shared__ signed char sA[CH * 64];
  __shared__ signed char sB[CH * 64];
  const int tid  = threadIdx.x;
  const int wave = tid >> 5, lane = tid & 31;
  const int l16  = lane & 15, hi = lane >> 4;
  const int mbase = blockIdx.y * CH;
  const int nbase = blockIdx.x * CH;
  const float wdq = fmaxf(*wsum * invcount, 1e-5f);

  v8i acc[8];
#pragma unroll
  for (int f = 0; f < 8; ++f)
#pragma unroll
    for (int r = 0; r < 8; ++r) acc[f][r] = 0;

  const int row = tid >> 1, half = tid & 1;
  for (int ks = 0; ks < DIMC; ks += 64) {
    {
      const signed char* ga = Aq + (size_t)(mbase + row) * DIMC + ks + half * 32;
      const signed char* gb = Wq + (size_t)(nbase + row) * DIMC + ks + half * 32;
      signed char* la = sA + row * 64 + half * 32;
      signed char* lb = sB + row * 64 + half * 32;
      async_copy_b128(la,      ga);
      async_copy_b128(la + 16, ga + 16);
      async_copy_b128(lb,      gb);
      async_copy_b128(lb + 16, gb + 16);
      if (ks + 64 < DIMC) __builtin_prefetch(ga + 64, 0, 1);
    }
    async_wait0();
    __syncthreads();

    v8i a;
    {
      const int* ap = (const int*)(sA + (wave * 16 + l16) * 64);
#pragma unroll
      for (int j = 0; j < 8; ++j) {
        int K = ((j >> 1) << 4) + ((j & 1) << 2) + (hi << 3);
        a[j] = ap[K >> 2];
      }
    }
    v8i bfr[8];
#pragma unroll
    for (int f = 0; f < 8; ++f) {
      const int* bp = (const int*)(sB + (f * 16 + l16) * 64);
#pragma unroll
      for (int v = 0; v < 8; ++v) {
        int K = ((v >> 1) << 4) + ((v & 1) << 2) + (hi << 3);
        bfr[f][v] = bp[K >> 2];
      }
    }
#pragma unroll
    for (int f = 0; f < 8; ++f)
      acc[f] = __builtin_amdgcn_wmma_i32_16x16x64_iu8(true, a, true, bfr[f], acc[f],
                                                      false, false);
    __syncthreads();
  }

#pragma unroll
  for (int f = 0; f < 8; ++f) {
    int n = nbase + f * 16 + l16;
    float bn = bias[n];
#pragma unroll
    for (int r = 0; r < 8; ++r) {
      int m = mbase + wave * 16 + r + (hi << 3);
      float val = (float)acc[f][r] * rowscale[m] * wdq + bn;
      if (outF) outF[(size_t)m * Ncols + n] = val;
      else      outH[(size_t)m * Ncols + n] = (_Float16)val;
    }
  }
}

__global__ void rope_pack_kernel(const _Float16* __restrict__ qkv,
                                 const float* __restrict__ cosb,
                                 const float* __restrict__ sinb,
                                 _Float16* __restrict__ qo,
                                 _Float16* __restrict__ ko,
                                 _Float16* __restrict__ vo) {
  size_t idx = (size_t)blockIdx.x * 256 + threadIdx.x;
  int dp = (int)(idx & 31);
  int h  = (int)((idx >> 5) & 15);
  size_t bn = idx >> 9;
  int n = (int)(bn & (SEQ - 1));
  int b = (int)(bn >> 13);

  const _Float16* rowp = qkv + bn * NQKV;
  float c = cosb[(size_t)n * 32 + dp];
  float s = sinb[(size_t)n * 32 + dp];
  int d0 = dp * 2;
  int base = h * HD + d0;
  float q0 = (float)rowp[base],            q1 = (float)rowp[base + 1];
  float k0 = (float)rowp[DIMC + base],     k1 = (float)rowp[DIMC + base + 1];
  float v0 = (float)rowp[2 * DIMC + base], v1 = (float)rowp[2 * DIMC + base + 1];

  size_t ob = ((((size_t)b * NH + h) * SEQ) + n) * HD + d0;
  qo[ob]     = (_Float16)(q0 * c - q1 * s);
  qo[ob + 1] = (_Float16)(q1 * c + q0 * s);
  ko[ob]     = (_Float16)(k0 * c - k1 * s);
  ko[ob + 1] = (_Float16)(k1 * c + k0 * s);
  vo[ob]     = (_Float16)v0;
  vo[ob + 1] = (_Float16)v1;
}

__global__ void attn_kernel(const _Float16* __restrict__ qg,
                            const _Float16* __restrict__ kg,
                            const _Float16* __restrict__ vg,
                            float* __restrict__ outg) {
  extern __shared__ char smem[];
  _Float16* sq    = (_Float16*)smem;
  _Float16* sk    = sq + CH * HD;
  _Float16* sv    = sk + CH * HD;
  _Float16* sp    = sv + CH * HD;
  float*    skv   = (float*)(sp + CH * CH);
  _Float16* skv16 = (_Float16*)(skv + HD * HD);

  const int bh = blockIdx.x;
  const int b = bh / NH, h = bh % NH;
  const int tid  = threadIdx.x;
  const int wave = tid >> 5, lane = tid & 31;
  const int l16  = lane & 15, hi = lane >> 4;
  const int m0   = wave * 16;

  const size_t headbase = ((size_t)b * NH + h) * (size_t)SEQ * HD;

  for (int i = tid; i < HD * HD; i += 256) { skv[i] = 0.f; skv16[i] = (_Float16)0.f; }
  __syncthreads();

  for (int c = 0; c < TCH; ++c) {
    {
      const size_t tb = headbase + (size_t)c * CH * HD;
#pragma unroll
      for (int i = 0; i < 4; ++i) {
        int eo = (tid + i * 256) * 8;
        async_copy_b128(sq + eo, qg + tb + eo);
        async_copy_b128(sk + eo, kg + tb + eo);
        async_copy_b128(sv + eo, vg + tb + eo);
      }
    }
    async_wait0();
    __syncthreads();

    v8f accS[8];
#pragma unroll
    for (int f = 0; f < 8; ++f)
#pragma unroll
      for (int r = 0; r < 8; ++r) accS[f][r] = 0.f;

    for (int ks = 0; ks < HD; ks += 32) {
      v16h a = load_a_rowmajor(sq, m0, ks, HD, l16, hi);
#pragma unroll
      for (int f = 0; f < 8; ++f) {
        v16h bkt = load_b_transposed(sk, f * 16, ks, HD, l16, hi);
        accS[f] = __builtin_amdgcn_wmma_f32_16x16x32_f16(false, a, false, bkt,
                                                         (short)0, accS[f], false, false);
      }
    }

    float pb[8][8], rmax[8], rsum[8];
#pragma unroll
    for (int r = 0; r < 8; ++r) { rmax[r] = NEG_INF; rsum[r] = 0.f; }
#pragma unroll
    for (int f = 0; f < 8; ++f)
#pragma unroll
      for (int r = 0; r < 8; ++r) {
        int nn = f * 16 + l16;
        int mr = m0 + r + (hi << 3);
        float vS = accS[f][r] * 0.125f;
        if (nn > mr) vS = NEG_INF;
        pb[f][r] = vS;
        rmax[r] = fmaxf(rmax[r], vS);
      }
#pragma unroll
    for (int r = 0; r < 8; ++r)
#pragma unroll
      for (int off = 1; off < 16; off <<= 1)
        rmax[r] = fmaxf(rmax[r], __shfl_xor(rmax[r], off, 32));
#pragma unroll
    for (int f = 0; f < 8; ++f)
#pragma unroll
      for (int r = 0; r < 8; ++r) {
        float e = __expf(pb[f][r] - rmax[r]);
        pb[f][r] = e; rsum[r] += e;
      }
#pragma unroll
    for (int r = 0; r < 8; ++r)
#pragma unroll
      for (int off = 1; off < 16; off <<= 1)
        rsum[r] += __shfl_xor(rsum[r], off, 32);
#pragma unroll
    for (int f = 0; f < 8; ++f)
#pragma unroll
      for (int r = 0; r < 8; ++r) {
        int mr = m0 + r + (hi << 3);
        sp[(size_t)mr * CH + f * 16 + l16] = (_Float16)(pb[f][r] / rsum[r]);
      }
    __syncthreads();

    v8f accO[4];
#pragma unroll
    for (int f = 0; f < 4; ++f)
#pragma unroll
      for (int r = 0; r < 8; ++r) accO[f][r] = 0.f;

    for (int ks = 0; ks < CH; ks += 32) {
      v16h a = load_a_rowmajor(sp, m0, ks, CH, l16, hi);
#pragma unroll
      for (int f = 0; f < 4; ++f) {
        v16h bv = load_b_rows(sv, f * 16, ks, HD, l16, hi);
        accO[f] = __builtin_amdgcn_wmma_f32_16x16x32_f16(false, a, false, bv,
                                                         (short)0, accO[f], false, false);
      }
    }
    for (int ks = 0; ks < HD; ks += 32) {
      v16h a = load_a_rowmajor(sq, m0, ks, HD, l16, hi);
#pragma unroll
      for (int f = 0; f < 4; ++f) {
        v16h bkv = load_b_rows(skv16, f * 16, ks, HD, l16, hi);
        accO[f] = __builtin_amdgcn_wmma_f32_16x16x32_f16(false, a, false, bkv,
                                                         (short)0, accO[f], false, false);
      }
    }

#pragma unroll
    for (int f = 0; f < 4; ++f)
#pragma unroll
      for (int r = 0; r < 8; ++r) {
        size_t nglob = (size_t)b * SEQ + (size_t)c * CH + m0 + r + (hi << 3);
        outg[nglob * DIMC + h * HD + f * 16 + l16] = accO[f][r];
      }
    __syncthreads();

    if (wave < 4) {
      const int d0 = wave * 16;
      v8f accKV[4];
#pragma unroll
      for (int f = 0; f < 4; ++f)
#pragma unroll
        for (int r = 0; r < 8; ++r) accKV[f][r] = 0.f;
      for (int ks = 0; ks < CH; ks += 32) {
        v16h a = load_a_colmajor(sk, d0, ks, HD, l16, hi);
#pragma unroll
        for (int f = 0; f < 4; ++f) {
          v16h bv = load_b_rows(sv, f * 16, ks, HD, l16, hi);
          accKV[f] = __builtin_amdgcn_wmma_f32_16x16x32_f16(false, a, false, bv,
                                                            (short)0, accKV[f], false, false);
        }
      }
#pragma unroll
      for (int f = 0; f < 4; ++f)
#pragma unroll
        for (int r = 0; r < 8; ++r) {
          int dd = d0 + r + (hi << 3);
          int ee = f * 16 + l16;
          float nv = skv[dd * HD + ee] + accKV[f][r];
          skv[dd * HD + ee]   = nv;
          skv16[dd * HD + ee] = (_Float16)nv;
        }
    }
    __syncthreads();
  }
}

extern "C" void kernel_launch(void* const* d_in, const int* in_sizes, int n_in,
                              void* d_out, int out_size, void* d_ws, size_t ws_size,
                              hipStream_t stream) {
  const float* x       = (const float*)d_in[0];
  const float* cosb    = (const float*)d_in[1];
  const float* sinb    = (const float*)d_in[2];
  const float* qkv_w   = (const float*)d_in[3];
  const float* qkv_b   = (const float*)d_in[4];
  const float* qkv_nw  = (const float*)d_in[5];
  const float* proj_w  = (const float*)d_in[6];
  const float* proj_b  = (const float*)d_in[7];
  const float* proj_nw = (const float*)d_in[8];
  const float* norm_w  = (const float*)d_in[9];

  char* w = (char*)d_ws;
  size_t off = 0;
  float* scal = (float*)(w + off);                 off += 256;
  signed char* xq1 = (signed char*)(w + off);      off += (size_t)MTOT * DIMC;
  float* rs1 = (float*)(w + off);                  off += (size_t)MTOT * 4;
  signed char* wq1 = (signed char*)(w + off);      off += (size_t)NQKV * DIMC;
  signed char* wq2 = (signed char*)(w + off);      off += (size_t)DIMC * DIMC;
  _Float16* qkvh = (_Float16*)(w + off);           off += (size_t)MTOT * NQKV * 2;
  _Float16* q16 = (_Float16*)(w + off);            off += (size_t)BB * NH * SEQ * HD * 2;
  _Float16* k16 = (_Float16*)(w + off);            off += (size_t)BB * NH * SEQ * HD * 2;
  _Float16* v16 = (_Float16*)(w + off);            off += (size_t)BB * NH * SEQ * HD * 2;
  float* attno = (float*)(w + off);                off += (size_t)MTOT * DIMC * 4;
  signed char* xq2 = (signed char*)(w + off);      off += (size_t)MTOT * DIMC;
  float* rs2 = (float*)(w + off);                  off += (size_t)MTOT * 4;

  (void)hipMemsetAsync(scal, 0, 256, stream);

  abssum_kernel<<<512, 256, 0, stream>>>(qkv_w, (size_t)NQKV * DIMC, &scal[0]);
  abssum_kernel<<<512, 256, 0, stream>>>(proj_w, (size_t)DIMC * DIMC, &scal[1]);
  wquant_kernel<<<2048, 256, 0, stream>>>(qkv_w, (size_t)NQKV * DIMC, &scal[0],
                                          1.f / (float)(NQKV * DIMC), wq1);
  wquant_kernel<<<1024, 256, 0, stream>>>(proj_w, (size_t)DIMC * DIMC, &scal[1],
                                          1.f / (float)(DIMC * DIMC), wq2);

  act_quant_kernel<<<MTOT, 256, 0, stream>>>(x, qkv_nw, RMS_EPS_BIT, nullptr, 0.f, xq1, rs1);
  gemm_iu8_kernel<<<dim3(NQKV / CH, MTOT / CH), 256, 0, stream>>>(
      xq1, rs1, wq1, &scal[0], 1.f / (float)(NQKV * DIMC), qkv_b, NQKV, nullptr, qkvh);

  rope_pack_kernel<<<(BB * SEQ * NH * 32) / 256, 256, 0, stream>>>(
      qkvh, cosb, sinb, q16, k16, v16);

  const int attn_lds = (3 * CH * HD + CH * CH) * 2 + HD * HD * 4 + HD * HD * 2;
  (void)hipFuncSetAttribute((const void*)attn_kernel,
                            hipFuncAttributeMaxDynamicSharedMemorySize, attn_lds);
  attn_kernel<<<BB * NH, 256, attn_lds, stream>>>(q16, k16, v16, attno);

  act_quant_kernel<<<MTOT, 256, 0, stream>>>(attno, norm_w, RMS_EPS_OUT,
                                             proj_nw, RMS_EPS_BIT, xq2, rs2);
  gemm_iu8_kernel<<<dim3(DIMC / CH, MTOT / CH), 256, 0, stream>>>(
      xq2, rs2, wq2, &scal[1], 1.f / (float)(DIMC * DIMC), proj_b, DIMC,
      (float*)d_out, nullptr);
}
